// TransDSAIndexer_6622839570904
// MI455X (gfx1250) — matrix operand order repaired and checked by reading, not compiled, not run on hardware
//
#include <hip/hip_runtime.h>
#include <hip/hip_bf16.h>
#include <math.h>

typedef __bf16 bf16_t;
typedef __bf16 bf16x8  __attribute__((ext_vector_type(8)));
typedef __bf16 bf16x16 __attribute__((ext_vector_type(16)));
typedef float  f32x8   __attribute__((ext_vector_type(8)));

#define S_LEN 2048
#define HIDD  2048
#define NH    16
#define NOPE_D 128
#define ROPE_D 64
#define VDIM_D 128
#define KVL   512
#define QLORA 1536
#define IDD   128
#define TOPK_N 1024
#define QKD   576              // KVL + ROPE
#define SCALE_F 0.08838834764831845f   // 128^-0.5
#define EPS_F 1e-6f

// ---------------- WMMA fragment helpers (wave32, bf16 16x16x32) -------------

__device__ __forceinline__ bf16x16 load_a_frag(const bf16_t* A, int ld, int lane) {
  // A points at tile origin (row0, k0). Lane m(0..15) holds row m.
  // lanes 0-15: K = 0..15, lanes 16-31: K = 16..31
  int m = lane & 15, hi = lane >> 4;
  return *(const bf16x16*)(A + (size_t)m * ld + hi * 16);
}

__device__ __forceinline__ bf16x16 load_b_frag(const bf16_t* Bt, int ld, int lane) {
  // Bt is N-major (Bt[n][k] == B[k][n]); tile origin (n0 row, k0 col).
  // lane n covers column n; lanes 0-15: K=0..15, lanes 16-31: K=16..31
  int n = lane & 15, hi = lane >> 4;
  return *(const bf16x16*)(Bt + (size_t)n * ld + hi * 16);
}

__device__ __forceinline__ f32x8 wmma_bf16(bf16x16 a, bf16x16 b, f32x8 c) {
  return __builtin_amdgcn_wmma_f32_16x16x32_bf16(false, a, false, b, (short)0, c, false, false);
}

// -------- generic bf16 GEMM, 16x64 per wave (A-frag reused 4x) --------------

__launch_bounds__(256)
__global__ void gemm_bf16_n64_k(const bf16_t* __restrict__ A, const bf16_t* __restrict__ Bt,
                                void* __restrict__ C, int M, int N, int K,
                                int lda, int ldb, int ldc,
                                long long sA, long long sB, long long sC, int outBf16) {
  int lane = threadIdx.x & 31;
  int wg = blockIdx.x * (blockDim.x >> 5) + (threadIdx.x >> 5);
  int ntn = N >> 6;
  if (wg >= (M >> 4) * ntn) return;
  int mi = wg / ntn, ni = wg % ntn;
  const bf16_t* Ab = A + (size_t)blockIdx.z * sA + (size_t)(mi * 16) * lda;
  const bf16_t* Bb = Bt + (size_t)blockIdx.z * sB + (size_t)(ni * 64) * ldb;
  f32x8 c0 = {}, c1 = {}, c2 = {}, c3 = {};
  for (int k = 0; k < K; k += 32) {
    bf16x16 a = load_a_frag(Ab + k, lda, lane);
    c0 = wmma_bf16(a, load_b_frag(Bb + (size_t) 0 * ldb + k, ldb, lane), c0);
    c1 = wmma_bf16(a, load_b_frag(Bb + (size_t)16 * ldb + k, ldb, lane), c1);
    c2 = wmma_bf16(a, load_b_frag(Bb + (size_t)32 * ldb + k, ldb, lane), c2);
    c3 = wmma_bf16(a, load_b_frag(Bb + (size_t)48 * ldb + k, ldb, lane), c3);
  }
  int n = lane & 15, hi = lane >> 4;
  size_t base = (size_t)blockIdx.z * sC;
#pragma unroll
  for (int g = 0; g < 8; g++) {
    size_t roff = base + (size_t)(mi * 16 + g + 8 * hi) * ldc + ni * 64 + n;
    if (outBf16) {
      ((bf16_t*)C)[roff +  0] = (bf16_t)c0[g];
      ((bf16_t*)C)[roff + 16] = (bf16_t)c1[g];
      ((bf16_t*)C)[roff + 32] = (bf16_t)c2[g];
      ((bf16_t*)C)[roff + 48] = (bf16_t)c3[g];
    } else {
      ((float*)C)[roff +  0] = c0[g];
      ((float*)C)[roff + 16] = c1[g];
      ((float*)C)[roff + 32] = c2[g];
      ((float*)C)[roff + 48] = c3[g];
    }
  }
}

// -------- narrow GEMM (16x16 per wave), for N not multiple of 64 ------------

__launch_bounds__(256)
__global__ void gemm_bf16_k(const bf16_t* __restrict__ A, const bf16_t* __restrict__ Bt,
                            void* __restrict__ C, int M, int N, int K,
                            int lda, int ldb, int ldc,
                            long long sA, long long sB, long long sC, int outBf16) {
  int lane = threadIdx.x & 31;
  int wg = blockIdx.x * (blockDim.x >> 5) + (threadIdx.x >> 5);
  int ntn = N >> 4;
  if (wg >= (M >> 4) * ntn) return;
  int mi = wg / ntn, ni = wg % ntn;
  const bf16_t* Ab = A + (size_t)blockIdx.z * sA + (size_t)(mi * 16) * lda;
  const bf16_t* Bb = Bt + (size_t)blockIdx.z * sB + (size_t)(ni * 16) * ldb;
  f32x8 c = {};
  for (int k = 0; k < K; k += 32)
    c = wmma_bf16(load_a_frag(Ab + k, lda, lane), load_b_frag(Bb + k, ldb, lane), c);
  int n = lane & 15, hi = lane >> 4;
  size_t base = (size_t)blockIdx.z * sC;
#pragma unroll
  for (int g = 0; g < 8; g++) {
    size_t off = base + (size_t)(mi * 16 + g + 8 * hi) * ldc + ni * 16 + n;
    if (outBf16) ((bf16_t*)C)[off] = (bf16_t)c[g];
    else         ((float*)C)[off]  = c[g];
  }
}

// ---------------- converts / packing ----------------------------------------

__global__ void cvt_bf16_k(const float* __restrict__ in, bf16_t* __restrict__ out, size_t n) {
  size_t i = (size_t)blockIdx.x * blockDim.x + threadIdx.x;
  if (i < n) out[i] = (bf16_t)in[i];
}

// kv_b_weight (H, 256, 512) -> k_bT (H,512,128) [transposed], v_b (H,128,512)
__global__ void pack_kvb_k(const float* __restrict__ kvb, bf16_t* __restrict__ kbT,
                           bf16_t* __restrict__ vb) {
  size_t i = (size_t)blockIdx.x * blockDim.x + threadIdx.x;
  if (i >= (size_t)NH * 128 * 512) return;
  int r = i % 512; int d = (i / 512) % 128; int h = i / (512 * 128);
  kbT[((size_t)h * 512 + r) * 128 + d] = (bf16_t)kvb[((size_t)h * 256 + d) * 512 + r];
  vb [((size_t)h * 128 + d) * 512 + r] = (bf16_t)kvb[((size_t)h * 256 + 128 + d) * 512 + r];
}

// kv[s][0..575] = concat(k_pass, k_rot); kvT[r][s] = k_pass[s][r]
__global__ void pack_kv_k(const float* __restrict__ kp, const float* __restrict__ kr,
                          bf16_t* __restrict__ kv, bf16_t* __restrict__ kvT) {
  size_t i = (size_t)blockIdx.x * blockDim.x + threadIdx.x;
  if (i >= (size_t)S_LEN * QKD) return;
  int d = i % QKD; int s = i / QKD;
  float v = (d < KVL) ? kp[(size_t)s * KVL + d] : kr[(size_t)s * ROPE_D + (d - KVL)];
  kv[i] = (bf16_t)v;
  if (d < KVL) kvT[(size_t)d * S_LEN + s] = (bf16_t)v;
}

// q_cat[h][s][512..575] = q_rot[h][s][0..63]
__global__ void pack_qrot_k(const float* __restrict__ qr, bf16_t* __restrict__ qcat) {
  size_t i = (size_t)blockIdx.x * blockDim.x + threadIdx.x;
  if (i >= (size_t)NH * S_LEN * ROPE_D) return;
  int d = i % ROPE_D; int s = (i / ROPE_D) % S_LEN; int h = i / (ROPE_D * S_LEN);
  qcat[((size_t)h * S_LEN + s) * QKD + KVL + d] = (bf16_t)qr[i];
}

// ---------------- index side: rope-interleave, rmsnorm, |w+b| ---------------

__global__ void build_index_k(const float* __restrict__ ql, const float* __restrict__ ckv,
                              const float* __restrict__ cosb, const float* __restrict__ sinb,
                              const float* __restrict__ knorm, float* __restrict__ wbuf,
                              const float* __restrict__ wbias,
                              bf16_t* __restrict__ iq, bf16_t* __restrict__ ik) {
  int s = blockIdx.x; int t = threadIdx.x;
  __shared__ float red[64];
  __shared__ float rinv;
  const float* cs = cosb + (size_t)s * ROPE_D;
  const float* sn = sinb + (size_t)s * ROPE_D;
  if (t < 64) { float x = ckv[(size_t)s * IDD + 64 + t]; red[t] = x * x; }
  __syncthreads();
  if (t == 0) { float v = 0; for (int i = 0; i < 64; i++) v += red[i];
                rinv = rsqrtf(v / 64.f + EPS_F); }
  __syncthreads();
  if (t < 64) {
    float x = ckv[(size_t)s * IDD + 64 + t];
    ik[(size_t)s * IDD + 64 + t] = (bf16_t)(x * rinv * knorm[t]);
  } else if (t < 96) {  // interleaved rope for index_k rot half
    int i = t - 64;
    float y0 = ckv[(size_t)s * IDD + 2 * i], y1 = ckv[(size_t)s * IDD + 2 * i + 1];
    ik[(size_t)s * IDD + i]      = (bf16_t)(y0 * cs[i] - y1 * sn[i]);
    ik[(size_t)s * IDD + 32 + i] = (bf16_t)(y1 * cs[32 + i] + y0 * sn[32 + i]);
  } else if (t < 112) { // |w + b|
    int h = t - 96;
    wbuf[(size_t)s * NH + h] = fabsf(wbuf[(size_t)s * NH + h] + wbias[h]);
  }
  for (int h = 0; h < NH; h++) {
    const float* q = ql + (size_t)s * (NH * IDD) + h * IDD;
    bf16_t* o = iq + ((size_t)s * NH + h) * IDD;
    if (t < 32) {
      float y0 = q[2 * t], y1 = q[2 * t + 1];
      o[t]      = (bf16_t)(y0 * cs[t] - y1 * sn[t]);
      o[32 + t] = (bf16_t)(y1 * cs[32 + t] + y0 * sn[32 + t]);
    } else if (t < 96) {
      int j = t - 32;
      o[64 + j] = (bf16_t)q[64 + j];
    }
  }
}

// ---------------- indexer score: sum_h w[q][h]*relu(scale*iq.ik) ------------

__launch_bounds__(256)
__global__ void idx_score_k(const bf16_t* __restrict__ iq, const bf16_t* __restrict__ ik,
                            const float* __restrict__ wbuf, float* __restrict__ score) {
  int qb = blockIdx.y * 16;
  int kb = blockIdx.x * 128;
  if (kb > qb + 15) return;               // entirely above diagonal (block-uniform)
  int lane = threadIdx.x & 31, wv = threadIdx.x >> 5;
  int n = lane & 15, hi = lane >> 4;
  __shared__ float wsh[16][16];
  wsh[threadIdx.x >> 4][threadIdx.x & 15] =
      wbuf[(size_t)(qb + (threadIdx.x >> 4)) * NH + (threadIdx.x & 15)];
  __syncthreads();
  int kbase = kb + wv * 16;
  float acc[8] = {0, 0, 0, 0, 0, 0, 0, 0};
  for (int h = 0; h < NH; h++) {
    f32x8 c = {};
    const bf16_t* Ab = iq + (size_t)qb * (NH * IDD) + h * IDD;
    const bf16_t* Bb = ik + (size_t)kbase * IDD;
#pragma unroll
    for (int k = 0; k < IDD; k += 32)
      c = wmma_bf16(load_a_frag(Ab + k, NH * IDD, lane), load_b_frag(Bb + k, IDD, lane), c);
#pragma unroll
    for (int g = 0; g < 8; g++) {
      float l = c[g] * SCALE_F;
      acc[g] += wsh[g + 8 * hi][h] * (l > 0.f ? l : 0.f);
    }
  }
#pragma unroll
  for (int g = 0; g < 8; g++) {
    int q = qb + g + 8 * hi, k = kbase + n;
    score[(size_t)q * S_LEN + k] = (k <= q) ? acc[g] : -INFINITY;
  }
}

// ---------------- exact top-k threshold: 4-pass radix select ----------------

__global__ void topk_thresh_k(const float* __restrict__ score, float* __restrict__ thr) {
  int q = blockIdx.x;
  if (q < TOPK_N) { if (threadIdx.x == 0) thr[q] = -INFINITY; return; }
  __shared__ unsigned hist[256];
  __shared__ unsigned sh_pref;
  __shared__ int sh_rank;
  const float* row = score + (size_t)q * S_LEN;
  int cnt = q + 1;
  if (threadIdx.x == 0) { sh_pref = 0u; sh_rank = TOPK_N; }
  unsigned maskh = 0u;
  for (int pass = 3; pass >= 0; pass--) {
    hist[threadIdx.x] = 0u;
    __syncthreads();
    unsigned pref = sh_pref;
    for (int k = threadIdx.x; k < cnt; k += 256) {
      unsigned v = __float_as_uint(row[k]);    // scores >= 0 -> bit-monotonic
      if ((v & maskh) == pref) atomicAdd(&hist[(v >> (pass * 8)) & 255u], 1u);
    }
    __syncthreads();
    if (threadIdx.x == 0) {
      int r = sh_rank; int b = 255;
      for (; b > 0; b--) { int c = (int)hist[b]; if (c >= r) break; r -= c; }
      sh_pref = pref | ((unsigned)b << (pass * 8));
      sh_rank = r;
    }
    maskh |= 0xFFu << (pass * 8);
    __syncthreads();
  }
  if (threadIdx.x == 0) thr[q] = __uint_as_float(sh_pref);
}

// ---------------- fused sparse flash attention + output projection ----------

__launch_bounds__(256)
__global__ void attn_k(const bf16_t* __restrict__ qcat, const bf16_t* __restrict__ kv,
                       const bf16_t* __restrict__ kvT, const bf16_t* __restrict__ vb,
                       const float* __restrict__ score, const float* __restrict__ thr,
                       float* __restrict__ out) {
  int h = blockIdx.y;
  int qb = blockIdx.x * 16;
  int lane = threadIdx.x & 31, wv = threadIdx.x >> 5;
  int n = lane & 15, hi = lane >> 4;

  __shared__ __align__(16) bf16_t Qs[16 * QKD];
  __shared__ __align__(16) float  Ss[16 * 128];
  __shared__ __align__(16) bf16_t Ps[16 * 128];
  __shared__ __align__(16) bf16_t Os[16 * 512];
  __shared__ float red[16 * 16];
  __shared__ float m_sh[16], l_sh[16], a_sh[16], inv_sh[16];

  const bf16_t* qsrc = qcat + ((size_t)h * S_LEN + qb) * QKD;  // 16 rows contiguous
  for (int i = threadIdx.x; i < 16 * QKD / 2; i += 256)
    ((unsigned int*)Qs)[i] = ((const unsigned int*)qsrc)[i];
  if (threadIdx.x < 16) { m_sh[threadIdx.x] = -INFINITY; l_sh[threadIdx.x] = 0.f; }
  __syncthreads();

  float thrv[8];
#pragma unroll
  for (int g = 0; g < 8; g++) thrv[g] = thr[qb + g + 8 * hi];

  f32x8 o0 = {}, o1 = {}, o2 = {}, o3 = {};
  int nblk = (qb + 16 + 127) >> 7;
  for (int blk = 0; blk < nblk; blk++) {
    int kb = blk << 7;
    // prefetch next key block's K/V panel lines (global_prefetch_b8)
    if (blk + 1 < nblk) {
      int kn = kb + 128;
      __builtin_prefetch(kv + (size_t)(kn + wv * 16 + n) * QKD + hi * 288, 0, 1);
      __builtin_prefetch(kvT + (size_t)(wv * 64 + hi * 32 + n) * S_LEN + kn, 0, 1);
    }
    // ---- logits: this wave's 16 keys over d=576 (18 WMMAs)
    f32x8 c = {};
    const bf16_t* Bb = kv + (size_t)(kb + wv * 16) * QKD;
#pragma unroll 6
    for (int k = 0; k < QKD; k += 32)
      c = wmma_bf16(load_a_frag(Qs + k, QKD, lane), load_b_frag(Bb + k, QKD, lane), c);
#pragma unroll
    for (int g = 0; g < 8; g++) {
      int m = g + 8 * hi, q = qb + m, k = kb + wv * 16 + n;
      float v = c[g] * SCALE_F;
      bool sel = (k <= q) && (score[(size_t)q * S_LEN + k] >= thrv[g]);
      Ss[m * 128 + wv * 16 + n] = sel ? v : -INFINITY;
    }
    __syncthreads();
    // ---- online softmax: row maxima
    {
      int row = threadIdx.x >> 4, j = threadIdx.x & 15;
      float mx = -INFINITY;
      for (int c2 = j * 8; c2 < j * 8 + 8; c2++) mx = fmaxf(mx, Ss[row * 128 + c2]);
      red[row * 16 + j] = mx;
    }
    __syncthreads();
    if (threadIdx.x < 16) {
      int row = threadIdx.x;
      float bm = -INFINITY;
      for (int j = 0; j < 16; j++) bm = fmaxf(bm, red[row * 16 + j]);
      float m_old = m_sh[row];
      float m_new = fmaxf(m_old, bm);
      float msafe = (m_new == -INFINITY) ? 0.f : m_new;
      float alpha = (m_old == -INFINITY) ? 0.f : __expf(m_old - msafe);
      m_sh[row] = m_new; a_sh[row] = alpha; l_sh[row] *= alpha;
    }
    __syncthreads();
    // ---- P = exp(S-m), row sums
    {
      int row = threadIdx.x >> 4, j = threadIdx.x & 15;
      float msafe = (m_sh[row] == -INFINITY) ? 0.f : m_sh[row];
      float ls = 0.f;
      for (int c2 = j * 8; c2 < j * 8 + 8; c2++) {
        float s0 = Ss[row * 128 + c2];
        float p = (s0 == -INFINITY) ? 0.f : __expf(s0 - msafe);
        Ps[row * 128 + c2] = (bf16_t)p;
        ls += p;
      }
      red[row * 16 + j] = ls;
    }
    __syncthreads();
    if (threadIdx.x < 16) {
      float sum = 0;
      for (int j = 0; j < 16; j++) sum += red[threadIdx.x * 16 + j];
      l_sh[threadIdx.x] += sum;
    }
    __syncthreads();
    // ---- O = alpha*O + P @ V  (wave owns r-chunk [wv*64, wv*64+64))
    {
      float al[8];
#pragma unroll
      for (int g = 0; g < 8; g++) al[g] = a_sh[g + 8 * hi];
#pragma unroll
      for (int g = 0; g < 8; g++) { o0[g] *= al[g]; o1[g] *= al[g]; o2[g] *= al[g]; o3[g] *= al[g]; }
#pragma unroll
      for (int k = 0; k < 128; k += 32) {
        bf16x16 a = load_a_frag(Ps + k, 128, lane);
        o0 = wmma_bf16(a, load_b_frag(kvT + (size_t)(wv * 64 +  0) * S_LEN + kb + k, S_LEN, lane), o0);
        o1 = wmma_bf16(a, load_b_frag(kvT + (size_t)(wv * 64 + 16) * S_LEN + kb + k, S_LEN, lane), o1);
        o2 = wmma_bf16(a, load_b_frag(kvT + (size_t)(wv * 64 + 32) * S_LEN + kb + k, S_LEN, lane), o2);
        o3 = wmma_bf16(a, load_b_frag(kvT + (size_t)(wv * 64 + 48) * S_LEN + kb + k, S_LEN, lane), o3);
      }
    }
    __syncthreads();
  }
  // ---- normalize, stage O, project with v_b
  if (threadIdx.x < 16)
    inv_sh[threadIdx.x] = (l_sh[threadIdx.x] > 0.f) ? 1.f / l_sh[threadIdx.x] : 0.f;
  __syncthreads();
#pragma unroll
  for (int g = 0; g < 8; g++) {
    int m = g + 8 * hi;
    float iv = inv_sh[m];
    Os[m * 512 + wv * 64 +  0 + n] = (bf16_t)(o0[g] * iv);
    Os[m * 512 + wv * 64 + 16 + n] = (bf16_t)(o1[g] * iv);
    Os[m * 512 + wv * 64 + 32 + n] = (bf16_t)(o2[g] * iv);
    Os[m * 512 + wv * 64 + 48 + n] = (bf16_t)(o3[g] * iv);
  }
  __syncthreads();
  {
    f32x8 c = {};
    const bf16_t* Bb = vb + ((size_t)h * VDIM_D + wv * 16) * KVL;  // v_b[h] is N-major (128x512)
#pragma unroll 4
    for (int k = 0; k < KVL; k += 32)
      c = wmma_bf16(load_a_frag(Os + k, 512, lane), load_b_frag(Bb + k, KVL, lane), c);
#pragma unroll
    for (int g = 0; g < 8; g++) {
      int m = g + 8 * hi;
      out[((size_t)(qb + m) * NH + h) * VDIM_D + wv * 16 + n] = c[g];
    }
  }
}

// ---------------------------------------------------------------------------

extern "C" void kernel_launch(void* const* d_in, const int* in_sizes, int n_in,
                              void* d_out, int out_size, void* d_ws, size_t ws_size,
                              hipStream_t stream) {
  const float* q_latent = (const float*)d_in[0];
  const float* hidden   = (const float*)d_in[1];
  const float* cosb     = (const float*)d_in[2];
  const float* sinb     = (const float*)d_in[3];
  const float* q_pass   = (const float*)d_in[4];
  const float* q_rot    = (const float*)d_in[5];
  const float* k_pass   = (const float*)d_in[6];
  const float* k_rot    = (const float*)d_in[7];
  const float* kvb_w    = (const float*)d_in[9];
  const float* wq_b_w   = (const float*)d_in[10];
  const float* wk_w     = (const float*)d_in[11];
  const float* knorm    = (const float*)d_in[12];
  const float* wproj_w  = (const float*)d_in[13];
  const float* wproj_b  = (const float*)d_in[14];
  float* out = (float*)d_out;

  char* p = (char*)d_ws;
  auto carve = [&](size_t bytes) -> char* {
    char* r = p; p += (bytes + 255) & ~(size_t)255; return r;
  };
  bf16_t* hidden_bf = (bf16_t*)carve((size_t)S_LEN * HIDD * 2);
  bf16_t* qlat_bf   = (bf16_t*)carve((size_t)S_LEN * QLORA * 2);
  bf16_t* wqb_bf    = (bf16_t*)carve((size_t)NH * IDD * QLORA * 2);
  bf16_t* wk_bf     = (bf16_t*)carve((size_t)IDD * HIDD * 2);
  bf16_t* wproj_bf  = (bf16_t*)carve((size_t)NH * HIDD * 2);
  bf16_t* qpass_bf  = (bf16_t*)carve((size_t)NH * S_LEN * NOPE_D * 2);
  bf16_t* kbT_bf    = (bf16_t*)carve((size_t)NH * KVL * NOPE_D * 2);
  bf16_t* vb_bf     = (bf16_t*)carve((size_t)NH * VDIM_D * KVL * 2);
  bf16_t* qcat_bf   = (bf16_t*)carve((size_t)NH * S_LEN * QKD * 2);
  bf16_t* kv_bf     = (bf16_t*)carve((size_t)S_LEN * QKD * 2);
  bf16_t* kvT_bf    = (bf16_t*)carve((size_t)KVL * S_LEN * 2);
  bf16_t* iq_bf     = (bf16_t*)carve((size_t)S_LEN * NH * IDD * 2);
  bf16_t* ik_bf     = (bf16_t*)carve((size_t)S_LEN * IDD * 2);
  float*  ql_f      = (float*)carve((size_t)S_LEN * NH * IDD * 4);
  float*  ckv_f     = (float*)carve((size_t)S_LEN * IDD * 4);
  float*  w_f       = (float*)carve((size_t)S_LEN * NH * 4);
  float*  score_f   = (float*)carve((size_t)S_LEN * S_LEN * 4);
  float*  thr_f     = (float*)carve((size_t)S_LEN * 4);

  auto blocks = [](size_t n) { return (unsigned)((n + 255) / 256); };

  // 1. converts to bf16
  cvt_bf16_k<<<blocks((size_t)S_LEN * HIDD), 256, 0, stream>>>(hidden, hidden_bf, (size_t)S_LEN * HIDD);
  cvt_bf16_k<<<blocks((size_t)S_LEN * QLORA), 256, 0, stream>>>(q_latent, qlat_bf, (size_t)S_LEN * QLORA);
  cvt_bf16_k<<<blocks((size_t)NH * IDD * QLORA), 256, 0, stream>>>(wq_b_w, wqb_bf, (size_t)NH * IDD * QLORA);
  cvt_bf16_k<<<blocks((size_t)IDD * HIDD), 256, 0, stream>>>(wk_w, wk_bf, (size_t)IDD * HIDD);
  cvt_bf16_k<<<blocks((size_t)NH * HIDD), 256, 0, stream>>>(wproj_w, wproj_bf, (size_t)NH * HIDD);
  cvt_bf16_k<<<blocks((size_t)NH * S_LEN * NOPE_D), 256, 0, stream>>>(q_pass, qpass_bf, (size_t)NH * S_LEN * NOPE_D);
  pack_kvb_k<<<blocks((size_t)NH * 128 * 512), 256, 0, stream>>>(kvb_w, kbT_bf, vb_bf);
  pack_kv_k<<<blocks((size_t)S_LEN * QKD), 256, 0, stream>>>(k_pass, k_rot, kv_bf, kvT_bf);
  pack_qrot_k<<<blocks((size_t)NH * S_LEN * ROPE_D), 256, 0, stream>>>(q_rot, qcat_bf);

  // 2. dense GEMMs (all WMMA)
  // ql = q_latent @ wq_b^T : (2048 x 2048), K=1536  -> 16x64 tiles
  gemm_bf16_n64_k<<<dim3((128 * 32) / 8, 1, 1), 256, 0, stream>>>(
      qlat_bf, wqb_bf, ql_f, S_LEN, NH * IDD, QLORA, QLORA, QLORA, NH * IDD, 0, 0, 0, 0);
  // ckv = hidden @ wk^T : (2048 x 128), K=2048  -> 16x64 tiles
  gemm_bf16_n64_k<<<dim3((128 * 2) / 8, 1, 1), 256, 0, stream>>>(
      hidden_bf, wk_bf, ckv_f, S_LEN, IDD, HIDD, HIDD, HIDD, IDD, 0, 0, 0, 0);
  // w = hidden @ wproj^T : (2048 x 16), K=2048  -> narrow path
  gemm_bf16_k<<<dim3((128 * 1) / 8, 1, 1), 256, 0, stream>>>(
      hidden_bf, wproj_bf, w_f, S_LEN, NH, HIDD, HIDD, HIDD, NH, 0, 0, 0, 0);
  // q_abs[h] = q_pass[h] @ k_b[h] : batched over heads, writes q_cat cols [0,512)
  gemm_bf16_n64_k<<<dim3((128 * 8) / 8, 1, NH), 256, 0, stream>>>(
      qpass_bf, kbT_bf, qcat_bf, S_LEN, KVL, NOPE_D, NOPE_D, NOPE_D, QKD,
      (long long)S_LEN * NOPE_D, (long long)KVL * NOPE_D, (long long)S_LEN * QKD, 1);

  // 3. index features (rope / rmsnorm / |w+b|)
  build_index_k<<<S_LEN, 128, 0, stream>>>(ql_f, ckv_f, cosb, sinb, knorm, w_f, wproj_b,
                                           iq_bf, ik_bf);

  // 4. indexer scores (causal, -inf above diagonal)
  idx_score_k<<<dim3(S_LEN / 128, S_LEN / 16, 1), 256, 0, stream>>>(iq_bf, ik_bf, w_f, score_f);

  // 5. exact top-1024 threshold per query row
  topk_thresh_k<<<S_LEN, 256, 0, stream>>>(score_f, thr_f);

  // 6. fused sparse flash attention + value up-projection
  attn_k<<<dim3(S_LEN / 16, NH, 1), 256, 0, stream>>>(qcat_bf, kv_bf, kvT_bf, vb_bf,
                                                      score_f, thr_f, out);
  (void)in_sizes; (void)n_in; (void)out_size; (void)ws_size;
}
